// maxcut_MPNN_13778255086000
// MI455X (gfx1250) — matrix operand order repaired and checked by reading, not compiled, not run on hardware
//
#include <hip/hip_runtime.h>
#include <hip/hip_bf16.h>
#include <cmath>

#define NN 50000
#define EE 800000
#define GG 50
#define DD 256
#define LLAYERS 4
#define BN_EPSF 1e-5f
#define SLOPEF 0.01f

typedef __bf16 bf16;
typedef __attribute__((ext_vector_type(16))) __bf16 v16bf;
typedef __attribute__((ext_vector_type(8)))  float  v8f;
typedef unsigned __attribute__((ext_vector_type(4))) u32x4;
typedef int      __attribute__((ext_vector_type(4))) i32x4;
typedef int      __attribute__((ext_vector_type(8))) i32x8;

#if defined(__AMDGCN__) && __has_builtin(__builtin_amdgcn_tensor_load_to_lds)
#define HAVE_TDM 1
#else
#define HAVE_TDM 0
#endif

__device__ __forceinline__ float leakyf(float v){ return v > 0.f ? v : SLOPEF * v; }

// monotone float<->uint mapping for atomic min/max over signed floats
__device__ __forceinline__ unsigned fmapu(float f){
  unsigned u = __float_as_uint(f);
  return (u & 0x80000000u) ? ~u : (u | 0x80000000u);
}
__device__ __forceinline__ float funmap(unsigned m){
  unsigned u = (m & 0x80000000u) ? (m ^ 0x80000000u) : ~m;
  return __uint_as_float(u);
}

// ---------------------------------------------------------------------------
// Weight prep: W (f32, 256 x NCOLS row-major) -> WT (bf16, NCOLS x 256 n-major)
// ---------------------------------------------------------------------------
__global__ void k_prepw(const float* __restrict__ W, bf16* __restrict__ WT, int ncols){
  int idx = blockIdx.x * blockDim.x + threadIdx.x;
  if (idx < 256 * ncols){
    int k = idx / ncols;
    int n = idx % ncols;
    WT[(size_t)n * 256 + k] = (bf16)W[idx];
  }
}

// ---------------------------------------------------------------------------
// WMMA GEMM: out = act(A(bf16, N x 256 row-major) @ W + bias)
// W pre-transposed as bf16 (NCOLS x 256).  Per-32-K chunk of W is staged into
// LDS by the Tensor Data Mover (2D tile, HW pad to LDT stride), waves then run
// v_wmma_f32_16x16x32_bf16 over NCOLS/16 column tiles.
// ---------------------------------------------------------------------------
template<int NCOLS, int ACT, bool OUTF, bool OUTB>
__global__ void __launch_bounds__(256) k_gemm(const bf16* __restrict__ A,
                                              const bf16* __restrict__ WTg,
                                              const float* __restrict__ bias,
                                              float* __restrict__ outF,
                                              bf16* __restrict__ outB,
                                              int nrows)
{
  constexpr int LDT = 40;              // padded LDS row stride (bf16 elems) = 80B
  constexpr int NT  = NCOLS / 16;
  __shared__ __attribute__((aligned(16))) bf16 WT[NCOLS * LDT];

  const int tid  = threadIdx.x;
  const int wave = tid >> 5;
  const int lane = tid & 31;
  const int m0   = (blockIdx.x * 8 + wave) * 16;
  const bool valid = (m0 < nrows);
  const int arow   = m0 + (lane & 15);
  const int koffA  = (lane < 16) ? 0 : 16;  // A frag: elems 0..15 = K koff..koff+15
  const int khalfB = (lane < 16) ? 0 : 16;  // B frag: elems 0..15 = K khalf..khalf+15

  v8f acc[NT];
  #pragma unroll
  for (int t = 0; t < NT; ++t){
    #pragma unroll
    for (int j = 0; j < 8; ++j) acc[t][j] = 0.0f;
  }

  for (int kc = 0; kc < 8; ++kc){
    const int kb = kc * 32;
#if HAVE_TDM
    // One TDM op stages the 2D weight tile (NCOLS rows x 64B) into LDS with
    // hardware padding of 16B per 64B row -> LDS row stride = 80B = LDT bf16.
    if (tid == 0){
      unsigned long long ga = (unsigned long long)(const void*)WTg
                            + (unsigned long long)kb * 2ull;
      unsigned ldsa = (unsigned)(size_t)(void*)WT;
      u32x4 g0 = { 1u,                                   // count=1, user mode
                   ldsa,                                 // lds_addr
                   (unsigned)ga,                         // global_addr[31:0]
                   (unsigned)((ga >> 32) & 0x01FFFFFFull) | 0x80000000u }; // addr[56:32] | type=2
      // g1: data_size=2B(code1), pad_enable, pad_interval=64B(code3), pad_amount=16B(code3)
      int w0 = (1 << 16) | (1 << 20) | (3 << 22) | (3 << 25);
      i32x8 g1 = { w0,
                   (int)(256u << 16),              // tensor_dim0[15:0] = 256 (<<16); abar=0
                   (int)((unsigned)NCOLS << 16),   // tensor_dim0[31:16]=0, tensor_dim1[15:0]=NCOLS
                   (int)(32u << 16),               // tensor_dim1[31:16]=0, tile_dim0=32
                   (int)NCOLS,                     // tile_dim1=NCOLS, tile_dim2=0
                   256,                            // tensor_dim0_stride[31:0] = 256 elems
                   0, 0 };
      i32x4 z4 = { 0, 0, 0, 0 };
#if __clang_major__ >= 23
      i32x8 z8 = { 0, 0, 0, 0, 0, 0, 0, 0 };
      __builtin_amdgcn_tensor_load_to_lds(g0, g1, z4, z4, z8, 0);
#else
      __builtin_amdgcn_tensor_load_to_lds(g0, g1, z4, z4, 0);
#endif
    }
    __builtin_amdgcn_s_wait_tensorcnt(0);   // no-op for non-issuing waves
#else
    // Fallback: cooperative bf16 copy into padded LDS layout
    for (int idx = tid; idx < NCOLS * 32; idx += 256){
      int n = idx >> 5, k = idx & 31;
      WT[n * LDT + k] = WTg[(size_t)n * 256 + kb + k];
    }
#endif
    __syncthreads();
    if (valid){
      const bf16* ap = A + (size_t)arow * DD + kb + koffA;
      union { v16bf v; uint4 u[2]; } af;
      af.u[0] = *(const uint4*)(ap);
      af.u[1] = *(const uint4*)(ap + 8);
      if (kc < 7) __builtin_prefetch(ap + 32, 0, 1);   // global_prefetch_b8
      #pragma unroll
      for (int t = 0; t < NT; ++t){
        const bf16* bp = &WT[(t * 16 + (lane & 15)) * LDT + khalfB];
        union { v16bf v; uint4 u[2]; } bfr;
        bfr.u[0] = *(const uint4*)(bp);
        bfr.u[1] = *(const uint4*)(bp + 8);
        acc[t] = __builtin_amdgcn_wmma_f32_16x16x32_bf16(
            false, af.v, false, bfr.v, (short)0, acc[t], false, false);
      }
    }
    __syncthreads();
  }

  if (valid){
    const int rofs = (lane >> 4) << 3;   // C/D: vgpr j -> M=j (lanes 0-15) / M=8+j (lanes 16-31)
    #pragma unroll
    for (int t = 0; t < NT; ++t){
      const int n = t * 16 + (lane & 15);
      const float bv = bias[n];
      #pragma unroll
      for (int j = 0; j < 8; ++j){
        const int r = m0 + j + rofs;
        float val = acc[t][j] + bv;
        if (ACT == 1) val = fmaxf(val, 0.0f);
        if (OUTF) outF[(size_t)r * NCOLS + n] = val;
        if (OUTB) outB[(size_t)r * NCOLS + n] = (bf16)val;
      }
    }
  }
}

// ---------------------------------------------------------------------------
// elementwise / segment kernels
// ---------------------------------------------------------------------------
__global__ void k_mv(const float* __restrict__ src, float* __restrict__ dst, int n){
  int i = blockIdx.x * blockDim.x + threadIdx.x;
  if (i < n) dst[i] = (fabsf(src[i]) > 0.0f) ? 1.0f : 0.0f;
}

__global__ void k_edge_or(const float* __restrict__ src, const int* __restrict__ row,
                          const int* __restrict__ col, float* __restrict__ dst){
  int e = blockIdx.x * blockDim.x + threadIdx.x;
  if (e < EE && fabsf(src[row[e]]) > 0.0f)
    atomicOr((unsigned int*)dst + col[e], 0x3F800000u);  // set to 1.0f
}

__global__ void k_deg(const int* __restrict__ row, float* __restrict__ deg){
  int e = blockIdx.x * blockDim.x + threadIdx.x;
  if (e < EE) atomicAdd(deg + row[e], 1.0f);
}

__global__ void k_agg_scalar(const float* __restrict__ x, const int* __restrict__ row,
                             const int* __restrict__ col, float* __restrict__ agg0){
  int e = blockIdx.x * blockDim.x + threadIdx.x;
  if (e < EE) atomicAdd(agg0 + col[e], x[row[e]]);
}

__global__ void k_outer1(const float* __restrict__ x, const float* __restrict__ agg0,
                         const float* __restrict__ eps1, const float* __restrict__ W1,
                         const float* __restrict__ b1, bf16* __restrict__ out){
  int idx = blockIdx.x * blockDim.x + threadIdx.x;
  if (idx < NN * DD){
    int i = idx >> 8, d = idx & 255;
    float u = (1.0f + eps1[0]) * x[i] + agg0[i];
    out[idx] = (bf16)fmaxf(u * W1[d] + b1[d], 0.0f);
  }
}

__global__ void k_edge_aggv(const int* __restrict__ row, const int* __restrict__ col,
                            const float* __restrict__ H, float* __restrict__ AGG){
  long long t = (long long)blockIdx.x * blockDim.x + threadIdx.x;
  int e = (int)(t >> 6);
  if (e >= EE) return;
  int c4 = ((int)t & 63) << 2;
  const float4 v = *(const float4*)(H + (size_t)row[e] * DD + c4);
  float* dst = AGG + (size_t)col[e] * DD + c4;
  atomicAdd(dst + 0, v.x); atomicAdd(dst + 1, v.y);
  atomicAdd(dst + 2, v.z); atomicAdd(dst + 3, v.w);
}

__global__ void k_mix(const float* __restrict__ H, const float* __restrict__ AGG,
                      const float* __restrict__ eps, int li, bf16* __restrict__ out){
  int idx = blockIdx.x * blockDim.x + threadIdx.x;
  if (idx < NN * DD) out[idx] = (bf16)((1.0f + eps[li]) * H[idx] + AGG[idx]);
}

__global__ void k_cast(const float* __restrict__ H, bf16* __restrict__ out){
  int idx = blockIdx.x * blockDim.x + threadIdx.x;
  if (idx < NN * DD) out[idx] = (bf16)H[idx];
}

__global__ void k_colstats(const float* __restrict__ V, float* __restrict__ s,
                           float* __restrict__ q, int nrows){
  int c = threadIdx.x;
  int r0 = blockIdx.x * 64;
  int r1 = r0 + 64 < nrows ? r0 + 64 : nrows;
  float ls = 0.f, lq = 0.f;
  for (int r = r0; r < r1; ++r){
    float v = V[(size_t)r * DD + c];
    ls += v; lq += v * v;
  }
  atomicAdd(s + c, ls); atomicAdd(q + c, lq);
}

__global__ void k_bnfinal(const float* __restrict__ s, const float* __restrict__ q,
                          const float* __restrict__ g, const float* __restrict__ b,
                          float* __restrict__ scale, float* __restrict__ shift){
  int c = threadIdx.x;
  float m   = s[c] * (1.0f / NN);
  float var = q[c] * (1.0f / NN) - m * m;
  float sc  = g[c] * rsqrtf(var + BN_EPSF);
  scale[c] = sc; shift[c] = b[c] - m * sc;
}

__global__ void k_post1(const float* __restrict__ V, const float* __restrict__ scale,
                        const float* __restrict__ shift, const float* __restrict__ mask,
                        float inv, float* __restrict__ HB){
  int idx = blockIdx.x * blockDim.x + threadIdx.x;
  if (idx < NN * DD){
    int i = idx >> 8, d = idx & 255;
    HB[idx] = leakyf(V[idx] * scale[d] + shift[d]) * mask[i] * inv;
  }
}

__global__ void k_post(const float* __restrict__ V, const float* __restrict__ scale,
                       const float* __restrict__ shift, const float* __restrict__ H,
                       const float* __restrict__ mask, float inv, float* __restrict__ HB){
  int idx = blockIdx.x * blockDim.x + threadIdx.x;
  if (idx < NN * DD){
    int i = idx >> 8, d = idx & 255;
    HB[idx] = (H[idx] + leakyf(V[idx] * scale[d] + shift[d])) * mask[i] * inv;
  }
}

__global__ void k_bnapply(const float* __restrict__ HB, const float* __restrict__ scale,
                          const float* __restrict__ shift, float* __restrict__ H){
  int idx = blockIdx.x * blockDim.x + threadIdx.x;
  if (idx < NN * DD){
    int d = idx & 255;
    H[idx] = HB[idx] * scale[d] + shift[d];
  }
}

__global__ void k_lin2(const float* __restrict__ Z, const float* __restrict__ mask,
                       const float* __restrict__ W2, const float* __restrict__ b2,
                       float* __restrict__ hfin){
  int i = blockIdx.x * blockDim.x + threadIdx.x;
  if (i < NN){
    float m = mask[i];
    float acc = 0.f;
    #pragma unroll
    for (int d = 0; d < 32; ++d) acc += leakyf(Z[(size_t)i * 32 + d]) * W2[d];
    hfin[i] = leakyf(acc * m + b2[0]) * m;
  }
}

__global__ void k_gminit(unsigned* gmax, unsigned* gmin){
  int g = threadIdx.x;
  if (g < GG){ gmax[g] = 0u; gmin[g] = 0xFFFFFFFFu; }
}

__global__ void k_segmm(const float* __restrict__ hfin, const int* __restrict__ batch,
                        unsigned* __restrict__ gmax, unsigned* __restrict__ gmin){
  int i = blockIdx.x * blockDim.x + threadIdx.x;
  if (i < NN){
    unsigned m = fmapu(hfin[i]);
    int g = batch[i];
    atomicMax(gmax + g, m);
    atomicMin(gmin + g, m);
  }
}

__global__ void k_probs(const float* __restrict__ hfin, const int* __restrict__ batch,
                        const unsigned* __restrict__ gmax, const unsigned* __restrict__ gmin,
                        float* __restrict__ probs){
  int i = blockIdx.x * blockDim.x + threadIdx.x;
  if (i < NN){
    int g = batch[i];
    float bmax = funmap(gmax[g]), bmin = funmap(gmin[g]);
    probs[i] = (hfin[i] - bmin) / (bmax + 1e-6f - bmin);
  }
}

__global__ void k_t1(const float* __restrict__ probs, const float* __restrict__ deg,
                     const int* __restrict__ batch, float* __restrict__ t1){
  int i = blockIdx.x * blockDim.x + threadIdx.x;
  if (i < NN) atomicAdd(t1 + batch[i], probs[i] * deg[i]);
}

__global__ void k_t2(const float* __restrict__ probs, const int* __restrict__ row,
                     const int* __restrict__ col, const int* __restrict__ batch,
                     float* __restrict__ t2){
  int e = blockIdx.x * blockDim.x + threadIdx.x;
  if (e < EE) atomicAdd(t2 + batch[row[e]], probs[row[e]] * probs[col[e]]);
}

__global__ void k_loss(const float* __restrict__ t1, const float* __restrict__ t2,
                       float* __restrict__ out){
  __shared__ float sh[64];
  int g = threadIdx.x;
  float l = 0.f;
  if (g < GG){ l = -(t1[g] - t2[g]); out[g] = l; }
  sh[g] = l;
  __syncthreads();
  if (g == 0){
    float s = 0.f;
    for (int k = 0; k < GG; ++k) s += sh[k];
    out[GG] = s / (float)GG;
  }
}

// ---------------------------------------------------------------------------
extern "C" void kernel_launch(void* const* d_in, const int* in_sizes, int n_in,
                              void* d_out, int out_size, void* d_ws, size_t ws_size,
                              hipStream_t stream)
{
  (void)in_sizes; (void)n_in; (void)out_size; (void)ws_size;
  const float* x    = (const float*)d_in[0];
  const int*   ei   = (const int*)d_in[1];
  const int*   row  = ei;
  const int*   col  = ei + EE;
  const int*   batch= (const int*)d_in[2];
  const float* c1W1 = (const float*)d_in[3];
  const float* c1b1 = (const float*)d_in[4];
  const float* c1W2 = (const float*)d_in[5];
  const float* c1b2 = (const float*)d_in[6];
  const float* c1bng= (const float*)d_in[7];
  const float* c1bnb= (const float*)d_in[8];
  const float* eps1 = (const float*)d_in[9];
  const float* cW1  = (const float*)d_in[10];
  const float* cb1  = (const float*)d_in[11];
  const float* cW2  = (const float*)d_in[12];
  const float* cb2  = (const float*)d_in[13];
  const float* cbng = (const float*)d_in[14];
  const float* cbnb = (const float*)d_in[15];
  const float* ceps = (const float*)d_in[16];
  const float* bn1g = (const float*)d_in[17];
  const float* bn1b = (const float*)d_in[18];
  const float* bnsg = (const float*)d_in[19];
  const float* bnsb = (const float*)d_in[20];
  const float* l1W  = (const float*)d_in[21];
  const float* l1b  = (const float*)d_in[22];
  const float* l2W  = (const float*)d_in[23];
  const float* l2b  = (const float*)d_in[24];

  // carve workspace (aligned 256B)
  char* p = (char*)d_ws;
  auto carve = [&](size_t bytes)->char*{
    char* r = p; p += (bytes + 255) & ~(size_t)255; return r;
  };
  float* H    = (float*)carve((size_t)NN * DD * 4);
  float* V    = (float*)carve((size_t)NN * DD * 4);
  float* AGG  = (float*)carve((size_t)NN * DD * 4);  // also reused as HB
  bf16*  BFA  = (bf16*) carve((size_t)NN * DD * 2);
  bf16*  BFB  = (bf16*) carve((size_t)NN * DD * 2);
  float* Z32  = (float*)carve((size_t)NN * 32 * 4);
  float* maskA= (float*)carve((size_t)NN * 4);
  float* maskB= (float*)carve((size_t)NN * 4);
  float* deg  = (float*)carve((size_t)NN * 4);
  float* agg0 = (float*)carve((size_t)NN * 4);
  float* hfin = (float*)carve((size_t)NN * 4);
  // pre-transposed bf16 weights (TDM source): 7 x 256x256 + 1 x 32x256
  bf16* Wt[8];
  for (int i = 0; i < 7; ++i) Wt[i] = (bf16*)carve((size_t)DD * DD * 2);
  Wt[7] = (bf16*)carve((size_t)32 * DD * 2);
  float* csum = (float*)carve(DD * 4);
  float* csq  = (float*)carve(DD * 4);
  float* sc1  = (float*)carve(DD * 4);
  float* sh1  = (float*)carve(DD * 4);
  float* sc2  = (float*)carve(DD * 4);
  float* sh2  = (float*)carve(DD * 4);
  unsigned* gmax = (unsigned*)carve(GG * 4);
  unsigned* gmin = (unsigned*)carve(GG * 4);
  float* t1 = (float*)carve(GG * 4);
  float* t2 = (float*)carve(GG * 4);

  const float inv = 1.0f / sqrtf((float)NN);
  dim3 B256(256);
  const int gN    = (NN + 255) / 256;
  const int gE    = (EE + 255) / 256;
  const int gND   = (NN * DD + 255) / 256;
  const int gEv   = (int)(((long long)EE * 64 + 255) / 256);
  const int gGemm = (NN + 127) / 128;
  const int gStat = (NN + 63) / 64;
  const int gW    = (DD * DD + 255) / 256;
  const int gW32  = (DD * 32 + 255) / 256;

  // weight prep (bf16, transposed; TDM-friendly)
  k_prepw<<<gW, B256, 0, stream>>>(c1W2, Wt[0], DD);
  for (int i = 0; i < 3; ++i){
    k_prepw<<<gW, B256, 0, stream>>>(cW1 + (size_t)i * DD * DD, Wt[1 + i], DD);
    k_prepw<<<gW, B256, 0, stream>>>(cW2 + (size_t)i * DD * DD, Wt[4 + i], DD);
  }
  k_prepw<<<gW32, B256, 0, stream>>>(l1W, Wt[7], 32);

  // stage 0: initial mask (from x) and degree
  k_mv<<<gN, B256, 0, stream>>>(x, maskA, NN);
  k_edge_or<<<gE, B256, 0, stream>>>(x, row, col, maskA);
  hipMemsetAsync(deg, 0, (size_t)NN * 4, stream);
  k_deg<<<gE, B256, 0, stream>>>(row, deg);

  // layer 1 (scalar input feature)
  hipMemsetAsync(agg0, 0, (size_t)NN * 4, stream);
  k_agg_scalar<<<gE, B256, 0, stream>>>(x, row, col, agg0);
  k_outer1<<<gND, B256, 0, stream>>>(x, agg0, eps1, c1W1, c1b1, BFA);
  k_gemm<256, 1, true, false><<<gGemm, B256, 0, stream>>>(BFA, Wt[0], c1b2, V, nullptr, NN);
  hipMemsetAsync(csum, 0, DD * 4, stream);
  hipMemsetAsync(csq,  0, DD * 4, stream);
  k_colstats<<<gStat, B256, 0, stream>>>(V, csum, csq, NN);
  k_bnfinal<<<1, B256, 0, stream>>>(csum, csq, c1bng, c1bnb, sc1, sh1);
  k_post1<<<gND, B256, 0, stream>>>(V, sc1, sh1, maskA, inv, AGG);
  hipMemsetAsync(csum, 0, DD * 4, stream);
  hipMemsetAsync(csq,  0, DD * 4, stream);
  k_colstats<<<gStat, B256, 0, stream>>>(AGG, csum, csq, NN);
  k_bnfinal<<<1, B256, 0, stream>>>(csum, csq, bn1g, bn1b, sc2, sh2);
  k_bnapply<<<gND, B256, 0, stream>>>(AGG, sc2, sh2, H);

  // layers 2..4
  float* mcur = maskA;
  float* mnext = maskB;
  for (int li = 0; li < LLAYERS - 1; ++li){
    hipMemsetAsync(AGG, 0, (size_t)NN * DD * 4, stream);
    k_edge_aggv<<<gEv, B256, 0, stream>>>(row, col, H, AGG);
    k_mix<<<gND, B256, 0, stream>>>(H, AGG, ceps, li, BFA);
    k_gemm<256, 1, false, true><<<gGemm, B256, 0, stream>>>(
        BFA, Wt[1 + li], cb1 + li * DD, nullptr, BFB, NN);
    k_gemm<256, 1, true, false><<<gGemm, B256, 0, stream>>>(
        BFB, Wt[4 + li], cb2 + li * DD, V, nullptr, NN);
    hipMemsetAsync(csum, 0, DD * 4, stream);
    hipMemsetAsync(csq,  0, DD * 4, stream);
    k_colstats<<<gStat, B256, 0, stream>>>(V, csum, csq, NN);
    k_bnfinal<<<1, B256, 0, stream>>>(csum, csq, cbng + li * DD, cbnb + li * DD, sc1, sh1);
    // mask BFS expansion (double buffered)
    k_mv<<<gN, B256, 0, stream>>>(mcur, mnext, NN);
    k_edge_or<<<gE, B256, 0, stream>>>(mcur, row, col, mnext);
    // residual + leaky(BN(v)) + mask scaling -> HB (reuse AGG)
    k_post<<<gND, B256, 0, stream>>>(V, sc1, sh1, H, mnext, inv, AGG);
    hipMemsetAsync(csum, 0, DD * 4, stream);
    hipMemsetAsync(csq,  0, DD * 4, stream);
    k_colstats<<<gStat, B256, 0, stream>>>(AGG, csum, csq, NN);
    k_bnfinal<<<1, B256, 0, stream>>>(csum, csq, bnsg + li * DD, bnsb + li * DD, sc2, sh2);
    k_bnapply<<<gND, B256, 0, stream>>>(AGG, sc2, sh2, H);
    float* tm = mcur; mcur = mnext; mnext = tm;
  }

  // head: lin1 (WMMA, 32 cols) then lin2 (dot-32 per node)
  k_cast<<<gND, B256, 0, stream>>>(H, BFA);
  k_gemm<32, 0, true, false><<<gGemm, B256, 0, stream>>>(BFA, Wt[7], l1b, Z32, nullptr, NN);
  k_lin2<<<gN, B256, 0, stream>>>(Z32, mcur, l2W, l2b, hfin);

  // per-graph min/max normalize -> probs directly into d_out
  k_gminit<<<1, 64, 0, stream>>>(gmax, gmin);
  k_segmm<<<gN, B256, 0, stream>>>(hfin, batch, gmax, gmin);
  float* probs = (float*)d_out;
  k_probs<<<gN, B256, 0, stream>>>(hfin, batch, gmax, gmin, probs);

  // loss terms and mean
  hipMemsetAsync(t1, 0, GG * 4, stream);
  hipMemsetAsync(t2, 0, GG * 4, stream);
  k_t1<<<gN, B256, 0, stream>>>(probs, deg, batch, t1);
  k_t2<<<gE, B256, 0, stream>>>(probs, row, col, batch, t2);
  k_loss<<<1, 64, 0, stream>>>(t1, t2, probs + NN);
}
